// GraphReasoningHead_84112639525597
// MI455X (gfx1250) — compile-verified, matrix operand order checked by reading, not run on hardware
//
#include <hip/hip_runtime.h>
#include <hip/hip_bf16.h>
#include <cstdint>
#include <cstddef>

// ---------------------------------------------------------------------------
// GraphReasoningHead for MI455X (gfx1250, wave32, WMMA).
// B=2, N=10000, E=160000, D=512, DC=8, L=3.
// GEMMs via v_wmma_f32_16x16x32_bf16 (fp32 accumulate); edge phase via wide
// vector loads + native fp32 global atomics; LayerNorm fused with residual.
// ---------------------------------------------------------------------------

typedef __bf16 bf16_t;
typedef __attribute__((ext_vector_type(16))) __bf16 v16bf;
typedef __attribute__((ext_vector_type(8)))  __bf16 v8bf;
typedef __attribute__((ext_vector_type(8)))  float  v8f;

constexpr int kB  = 2;
constexpr int kN  = 10000;
constexpr int kE  = 160000;
constexpr int kD  = 512;
constexpr int kDC = 8;
constexpr int kL  = 3;
constexpr float kEps = 1e-5f;

// GEMM tiling
constexpr int TM  = 128;
constexpr int TN  = 128;
constexpr int TK  = 32;
constexpr int LDK = 40;   // padded halves per LDS row (80 B -> 16B-aligned frags)

// ---------------------------------------------------------------------------
// Utility: zero fill
// ---------------------------------------------------------------------------
__global__ void fill0_kernel(float* __restrict__ p, size_t n) {
  size_t i = (size_t)blockIdx.x * blockDim.x + threadIdx.x;
  size_t stride = (size_t)gridDim.x * blockDim.x;
  for (; i < n; i += stride) p[i] = 0.0f;
}

// ---------------------------------------------------------------------------
// coords = x @ W_coord   (per-node GEMV, K=512, DC=8; W_coord is L0-resident)
// ---------------------------------------------------------------------------
__global__ __launch_bounds__(256) void coords_kernel(
    const float* __restrict__ x, const float* __restrict__ Wc,
    float* __restrict__ coords, int total) {
  int i = blockIdx.x * blockDim.x + threadIdx.x;
  if (i >= total) return;
  const float* row = x + (size_t)i * kD;
  float acc[kDC];
#pragma unroll
  for (int c = 0; c < kDC; ++c) acc[c] = 0.0f;
  for (int k = 0; k < kD; k += 4) {
    float4 f = *(const float4*)(row + k);
#pragma unroll
    for (int c = 0; c < kDC; ++c) {
      acc[c] += f.x * Wc[(k + 0) * kDC + c];
      acc[c] += f.y * Wc[(k + 1) * kDC + c];
      acc[c] += f.z * Wc[(k + 2) * kDC + c];
      acc[c] += f.w * Wc[(k + 3) * kDC + c];
    }
  }
#pragma unroll
  for (int c = 0; c < kDC; ++c) coords[(size_t)i * kDC + c] = acc[c];
}

// ---------------------------------------------------------------------------
// transport[b,e] = exp(-||coords[b,dst]-coords[b,src]||)
// ---------------------------------------------------------------------------
__global__ __launch_bounds__(256) void transport_kernel(
    const float* __restrict__ coords, const int* __restrict__ ei,
    float* __restrict__ tr) {
  int i = blockIdx.x * blockDim.x + threadIdx.x;
  if (i >= kB * kE) return;
  int b = i / kE;
  int e = i - b * kE;
  int s = ei[e];
  int d = ei[kE + e];
  const float* cs = coords + ((size_t)b * kN + s) * kDC;
  const float* cd = coords + ((size_t)b * kN + d) * kDC;
  float acc = 0.0f;
#pragma unroll
  for (int c = 0; c < kDC; ++c) {
    float df = cd[c] - cs[c];
    acc += df * df;
  }
  tr[i] = expf(-sqrtf(acc));
}

// ---------------------------------------------------------------------------
// deg[dst] += 1  (native fp32 atomic at L2)
// ---------------------------------------------------------------------------
__global__ __launch_bounds__(256) void degree_kernel(
    const int* __restrict__ ei, float* __restrict__ deg) {
  int e = blockIdx.x * blockDim.x + threadIdx.x;
  if (e >= kE) return;
  int d = ei[kE + e];
  __hip_atomic_fetch_add(&deg[d], 1.0f, __ATOMIC_RELAXED, __HIP_MEMORY_SCOPE_AGENT);
}

// ---------------------------------------------------------------------------
// Wt[n*D + k] = bf16(W[k*D + n])  -- transpose + downconvert (1 MB, trivial)
// ---------------------------------------------------------------------------
__global__ __launch_bounds__(256) void convw_kernel(
    const float* __restrict__ W, bf16_t* __restrict__ Wt) {
  int idx = blockIdx.x * blockDim.x + threadIdx.x;  // k*D + n, coalesced read
  int k = idx / kD;
  int n = idx - k * kD;
  Wt[(size_t)n * kD + k] = (bf16_t)W[idx];
}

// ---------------------------------------------------------------------------
// WMMA GEMM: C[M x 512] = A[M x 512](fp32) * W[512 x 512] + bias
// Bt is pre-transposed bf16: Bt[n*512 + k] = W[k,n].
// Block = 256 threads (8 waves), tile 128x128, K-step 32.
// Wave w: M-offset 32*(w&3), N-offset 64*(w>>2); 2x4 grid of 16x16 WMMA tiles.
// ---------------------------------------------------------------------------
__device__ inline v16bf load_frag_lds(const bf16_t* __restrict__ p, int kb) {
  // p -> start of this row's K-slab in LDS; kb in {0,8}
  v8bf lo = *(const v8bf*)(p + kb);        // K = kb..kb+7
  v8bf hi = *(const v8bf*)(p + 16 + kb);   // K = 16+kb..16+kb+7
  return __builtin_shufflevector(lo, hi, 0, 1, 2, 3, 4, 5, 6, 7,
                                 8, 9, 10, 11, 12, 13, 14, 15);
}

__global__ __launch_bounds__(256) void gemm_bf16_wmma_kernel(
    const float* __restrict__ A, const bf16_t* __restrict__ Bt,
    const float* __restrict__ bias, float* __restrict__ C, int Mtot) {
  __shared__ bf16_t lA[TM * LDK];
  __shared__ bf16_t lB[TN * LDK];

  const int m0   = blockIdx.x * TM;
  const int n0   = blockIdx.y * TN;
  const int tid  = threadIdx.x;
  const int lane = tid & 31;
  const int wave = tid >> 5;
  const int wm   = (wave & 3) * 32;   // wave M offset within tile
  const int wn   = (wave >> 2) * 64;  // wave N offset within tile

  v8f zero8 = {};
  v8f acc[2][4];
#pragma unroll
  for (int i = 0; i < 2; ++i)
#pragma unroll
    for (int j = 0; j < 4; ++j) acc[i][j] = zero8;

  const int r_ld  = tid >> 1;          // 0..127: row (A) / col (Bt) to load
  const int kk_ld = (tid & 1) * 16;    // 16-element K half-slab

  for (int k0 = 0; k0 < kD; k0 += TK) {
    // ---- A tile: fp32 -> bf16 into LDS (zero-fill rows past M) ----
    {
      int gm = m0 + r_ld;
      bf16_t* dstp = &lA[r_ld * LDK + kk_ld];
      if (gm < Mtot) {
        const float* srcp = A + (size_t)gm * kD + k0 + kk_ld;
        if (k0 + TK < kD) __builtin_prefetch(srcp + TK, 0, 0);  // global_prefetch
#pragma unroll
        for (int q = 0; q < 4; ++q) {
          float4 f = ((const float4*)srcp)[q];
          dstp[q * 4 + 0] = (bf16_t)f.x;
          dstp[q * 4 + 1] = (bf16_t)f.y;
          dstp[q * 4 + 2] = (bf16_t)f.z;
          dstp[q * 4 + 3] = (bf16_t)f.w;
        }
      } else {
        bf16_t z = (bf16_t)0.0f;
#pragma unroll
        for (int q = 0; q < 16; ++q) dstp[q] = z;
      }
    }
    // ---- Bt tile: bf16 -> LDS (pure copy, 32 B per thread) ----
    {
      const bf16_t* srcp = Bt + (size_t)(n0 + r_ld) * kD + k0 + kk_ld;
      bf16_t* dstp = &lB[r_ld * LDK + kk_ld];
      ((uint4*)dstp)[0] = ((const uint4*)srcp)[0];
      ((uint4*)dstp)[1] = ((const uint4*)srcp)[1];
    }
    __syncthreads();

    // ---- fragments (per ISA 16-bit 16x32 layout) ----
    const int rr = lane & 15;
    const int kb = (lane >> 4) << 3;  // 0 or 8
    v16bf af[2], bfv[4];
#pragma unroll
    for (int i = 0; i < 2; ++i)
      af[i] = load_frag_lds(&lA[(wm + 16 * i + rr) * LDK], kb);
#pragma unroll
    for (int j = 0; j < 4; ++j)
      bfv[j] = load_frag_lds(&lB[(wn + 16 * j + rr) * LDK], kb);

#pragma unroll
    for (int i = 0; i < 2; ++i)
#pragma unroll
      for (int j = 0; j < 4; ++j)
        acc[i][j] = __builtin_amdgcn_wmma_f32_16x16x32_bf16(
            false, af[i], false, bfv[j], (short)0, acc[i][j], false, false);
    __syncthreads();
  }

  // ---- store D-matrix (+bias); 16x16 f32 layout: lane n = l&15, M = r + 8*(l>>4)
  const int nlane = lane & 15;
  const int mhalf = (lane >> 4) * 8;
#pragma unroll
  for (int i = 0; i < 2; ++i) {
#pragma unroll
    for (int j = 0; j < 4; ++j) {
      int gn = n0 + wn + 16 * j + nlane;
      float bv = bias[gn];
#pragma unroll
      for (int r = 0; r < 8; ++r) {
        int gm = m0 + wm + 16 * i + mhalf + r;
        if (gm < Mtot) C[(size_t)gm * kD + gn] = acc[i][j][r] + bv;
      }
    }
  }
}

// ---------------------------------------------------------------------------
// Edge phase: msg = x_field[b,src]*transport[b,e] + edge_attr[e]
//             upd[b,dst] += msg (atomic);  optionally emit msg (last layer)
// One block per edge, 128 threads x float4 covers D=512 for both batches.
// ---------------------------------------------------------------------------
__global__ __launch_bounds__(128) void edge_kernel(
    const float* __restrict__ xf, const float* __restrict__ ea,
    const float* __restrict__ transport, const int* __restrict__ ei,
    float* __restrict__ upd, float* __restrict__ edge_out, int write_edge) {
  int e = blockIdx.x;
  int c = threadIdx.x;  // float4 index, 0..127
  int src = ei[e];
  int dst = ei[kE + e];
  float4 eav = ((const float4*)(ea + (size_t)e * kD))[c];
#pragma unroll
  for (int b = 0; b < kB; ++b) {
    float t = transport[(size_t)b * kE + e];
    float4 xv = ((const float4*)(xf + ((size_t)b * kN + src) * kD))[c];
    float4 m;
    m.x = xv.x * t + eav.x;
    m.y = xv.y * t + eav.y;
    m.z = xv.z * t + eav.z;
    m.w = xv.w * t + eav.w;
    float* up = upd + ((size_t)b * kN + dst) * kD + c * 4;
    __hip_atomic_fetch_add(up + 0, m.x, __ATOMIC_RELAXED, __HIP_MEMORY_SCOPE_AGENT);
    __hip_atomic_fetch_add(up + 1, m.y, __ATOMIC_RELAXED, __HIP_MEMORY_SCOPE_AGENT);
    __hip_atomic_fetch_add(up + 2, m.z, __ATOMIC_RELAXED, __HIP_MEMORY_SCOPE_AGENT);
    __hip_atomic_fetch_add(up + 3, m.w, __ATOMIC_RELAXED, __HIP_MEMORY_SCOPE_AGENT);
    if (write_edge)
      ((float4*)(edge_out + ((size_t)b * kE + e) * kD))[c] = m;
  }
}

// ---------------------------------------------------------------------------
// LayerNorm (optionally fused with x += upd/deg). One block per (b,n) row.
// 256 threads x 2 elements; block reduction through LDS.
// ---------------------------------------------------------------------------
template <bool WITH_UPD>
__global__ __launch_bounds__(256) void ln_kernel(
    const float* __restrict__ xin, const float* __restrict__ upd,
    const float* __restrict__ deg, const float* __restrict__ g,
    const float* __restrict__ beta, float* __restrict__ xout) {
  int row = blockIdx.x;  // b*N + n
  int t = threadIdx.x;
  const float* xr = xin + (size_t)row * kD;
  float v0 = xr[t];
  float v1 = xr[t + 256];
  if (WITH_UPD) {
    int n = row % kN;
    float invdeg = 1.0f / fmaxf(deg[n], 1.0f);
    const float* ur = upd + (size_t)row * kD;
    v0 += ur[t] * invdeg;
    v1 += ur[t + 256] * invdeg;
  }
  __shared__ float red[256];
  const float invD = 1.0f / (float)kD;
  red[t] = v0 + v1;
  __syncthreads();
  for (int off = 128; off > 0; off >>= 1) {
    if (t < off) red[t] += red[t + off];
    __syncthreads();
  }
  float mean = red[0] * invD;
  __syncthreads();
  float d0 = v0 - mean, d1 = v1 - mean;
  red[t] = d0 * d0 + d1 * d1;
  __syncthreads();
  for (int off = 128; off > 0; off >>= 1) {
    if (t < off) red[t] += red[t + off];
    __syncthreads();
  }
  float rstd = rsqrtf(red[0] * invD + kEps);
  xout[(size_t)row * kD + t]       = d0 * rstd * g[t] + beta[t];
  xout[(size_t)row * kD + t + 256] = d1 * rstd * g[t + 256] + beta[t + 256];
}

// ---------------------------------------------------------------------------
// Host orchestration
// ---------------------------------------------------------------------------
extern "C" void kernel_launch(void* const* d_in, const int* in_sizes, int n_in,
                              void* d_out, int out_size, void* d_ws, size_t ws_size,
                              hipStream_t stream) {
  (void)in_sizes; (void)n_in; (void)out_size; (void)ws_size;
  const float* node_features = (const float*)d_in[0];
  const int*   edge_index    = (const int*)d_in[1];
  const float* edge_attr     = (const float*)d_in[2];
  const float* W_coord       = (const float*)d_in[3];
  const float* W_field       = (const float*)d_in[4];
  const float* b_field       = (const float*)d_in[5];
  const float* ln_g          = (const float*)d_in[6];
  const float* ln_b          = (const float*)d_in[7];
  const float* out_ln_g      = (const float*)d_in[8];
  const float* out_ln_b      = (const float*)d_in[9];
  const float* W_out         = (const float*)d_in[10];
  const float* b_out         = (const float*)d_in[11];

  float* out      = (float*)d_out;
  float* node_out = out;                              // (B, N, D)
  float* edge_out = out + (size_t)kB * kN * kD;       // (B, E, D)

  // Workspace carve-up (~125 MB total)
  uint8_t* wp = (uint8_t*)d_ws;
  auto alloc = [&](size_t bytes) -> void* {
    void* p = (void*)wp;
    wp += (bytes + 255) & ~(size_t)255;
    return p;
  };
  float*  coords    = (float*)alloc(sizeof(float) * (size_t)kB * kN * kDC);
  float*  transport = (float*)alloc(sizeof(float) * (size_t)kB * kE);
  float*  deg       = (float*)alloc(sizeof(float) * (size_t)kN);
  float*  x         = (float*)alloc(sizeof(float) * (size_t)kB * kN * kD);
  float*  xf        = (float*)alloc(sizeof(float) * (size_t)kB * kN * kD);
  float*  upd       = (float*)alloc(sizeof(float) * (size_t)kB * kN * kD);
  bf16_t* Wt        = (bf16_t*)alloc(sizeof(bf16_t) * (size_t)kD * kD);

  const int M = kB * kN;  // 20000 GEMM rows

  // x <- node_features (fresh every call; buffers are re-initialized below)
  hipMemcpyAsync(x, node_features, sizeof(float) * (size_t)M * kD,
                 hipMemcpyDeviceToDevice, stream);

  coords_kernel<<<(M + 255) / 256, 256, 0, stream>>>(node_features, W_coord, coords, M);
  transport_kernel<<<(kB * kE + 255) / 256, 256, 0, stream>>>(coords, edge_index, transport);
  fill0_kernel<<<64, 256, 0, stream>>>(deg, (size_t)kN);
  degree_kernel<<<(kE + 255) / 256, 256, 0, stream>>>(edge_index, deg);

  dim3 ggrid((M + TM - 1) / TM, kD / TN);  // (157, 4)
  for (int l = 0; l < kL; ++l) {
    convw_kernel<<<(kD * kD) / 256, 256, 0, stream>>>(W_field + (size_t)l * kD * kD, Wt);
    gemm_bf16_wmma_kernel<<<ggrid, 256, 0, stream>>>(x, Wt, b_field + l * kD, xf, M);
    fill0_kernel<<<2048, 256, 0, stream>>>(upd, (size_t)M * kD);
    edge_kernel<<<kE, 128, 0, stream>>>(xf, edge_attr, transport, edge_index, upd,
                                        edge_out, (l == kL - 1) ? 1 : 0);
    ln_kernel<true><<<M, 256, 0, stream>>>(x, upd, deg, ln_g + l * kD, ln_b + l * kD, x);
  }
  // output projection: LN then GEMM with W_out
  ln_kernel<false><<<M, 256, 0, stream>>>(x, nullptr, nullptr, out_ln_g, out_ln_b, xf);
  convw_kernel<<<(kD * kD) / 256, 256, 0, stream>>>(W_out, Wt);
  gemm_bf16_wmma_kernel<<<ggrid, 256, 0, stream>>>(xf, Wt, b_out, node_out, M);
}
